// IaBNet_with_affinity_71519795413303
// MI455X (gfx1250) — matrix operand order repaired and checked by reading, compiled, not run on hardware
//
#include <hip/hip_runtime.h>
#include <hip/hip_bf16.h>

#define E_     128
#define C_     128
#define HEADS_ 4
#define HC_    32
#define S_     5
#define B_     2
#define NP_    512
#define NC_    128
#define RZ_    (B_*NP_*NC_)   /* 131072 rows of z  */
#define RPP_   (B_*NP_*NP_)   /* 524288 rows of pp */
#define RCP_   (B_*NC_*NC_)   /* 32768 rows of cp  */
#define LDL_   144            /* LDS f16 row stride (pad) */

typedef __attribute__((ext_vector_type(16))) _Float16 v16h;
typedef __attribute__((ext_vector_type(8)))  _Float16 v8h;
typedef __attribute__((ext_vector_type(8)))  float    v8f;

__device__ inline v8f v8f_zero() {
    v8f z = {0.f,0.f,0.f,0.f,0.f,0.f,0.f,0.f};
    return z;
}

// hardware v_rcp_f32 (avoid IEEE division microcode: div_scale + NR + fixup)
__device__ inline float fast_rcp(float x) { return __builtin_amdgcn_rcpf(x); }
__device__ inline float sigmoidf_(float x) { return fast_rcp(1.f + __expf(-x)); }

// ---- WMMA fragment loaders (contiguous k = 16h + s, same order in A and B, wave32) ----
// A 16x32 f16: lanes 0-15 -> M=lane, K = 0..15 ; lanes 16-31 -> K = 16..31
__device__ inline v16h frag_a(const _Float16* A, int lda, int k0, int lane) {
    const int m  = lane & 15;
    const int kb = (lane & 16) ? 16 : 0;
    const _Float16* p = A + (size_t)m * lda + k0 + kb;
    v8h lo = *(const v8h*)(p);
    v8h hi = *(const v8h*)(p + 8);
    v16h r;
#pragma unroll
    for (int i = 0; i < 8; ++i) { r[i] = lo[i]; r[i + 8] = hi[i]; }
    return r;
}
// B 32x16 f16 loaded from B^T [N,K]: lane n holds K=0..15 (lanes 0-15) or 16..31 (lanes 16-31)
__device__ inline v16h frag_bT(const _Float16* BT, size_t ldb, int n0, int k0, int lane) {
    const int n  = n0 + (lane & 15);
    const int kb = (lane & 16) ? 16 : 0;
    return *(const v16h*)(BT + (size_t)n * ldb + k0 + kb);
}
__device__ inline v8f wmma32(v16h a, v16h b, v8f acc) {
    return __builtin_amdgcn_wmma_f32_16x16x32_f16(false, a, false, b, (short)0, acc, false, false);
}
__device__ inline v8f gemm_tile(const _Float16* A, int lda, const _Float16* BT, size_t ldb,
                                int n0, int K, int lane) {
    v8f acc = v8f_zero();
    for (int k0 = 0; k0 < K; k0 += 32)
        acc = wmma32(frag_a(A, lda, k0, lane), frag_bT(BT, ldb, n0, k0, lane), acc);
    return acc;
}

// ---- LayerNorm(128) of a 16-row tile into f16 LDS tile (stride LDL_) ----
template <typename T>
__device__ inline void ln_tile_to_lds(const T* X, size_t r0, const float* g, const float* b,
                                      _Float16* lds, int t) {
    const int m = t >> 3, sub = t & 7;   // 8 lanes per row, 16 cols each
    const T* xr = X + (r0 + m) * 128;
    float v[16]; float s = 0.f;
#pragma unroll
    for (int i = 0; i < 16; ++i) { v[i] = (float)xr[sub * 16 + i]; s += v[i]; }
#pragma unroll
    for (int o = 1; o < 8; o <<= 1) s += __shfl_xor(s, o, 32);
    const float mean = s * (1.f / 128.f);
    float q = 0.f;
#pragma unroll
    for (int i = 0; i < 16; ++i) { float d = v[i] - mean; q += d * d; }
#pragma unroll
    for (int o = 1; o < 8; o <<= 1) q += __shfl_xor(q, o, 32);
    const float rstd = rsqrtf(q * (1.f / 128.f) + 1e-5f);
#pragma unroll
    for (int i = 0; i < 16; ++i) {
        int col = sub * 16 + i;
        lds[m * LDL_ + col] = (_Float16)((v[i] - mean) * rstd * g[col] + b[col]);
    }
}

// =================== kernels ===================

__global__ void k_transpose_f16(const float* __restrict__ src, _Float16* __restrict__ dst,
                                int K, int N) {
    int idx = blockIdx.x * blockDim.x + threadIdx.x;
    if (idx >= K * N) return;
    int n = idx % N, k = idx / N;
    dst[(size_t)n * K + k] = (_Float16)src[(size_t)k * N + n];
}

__global__ void k_pair_emb(const float* __restrict__ coords, const float* __restrict__ W,
                           const float* __restrict__ bias, _Float16* __restrict__ out, int N) {
    const int pid = blockIdx.x;            // b*N*N + i*N + j
    const int c = threadIdx.x;             // 128 channels
    const int b = pid / (N * N);
    const int rem = pid % (N * N);
    const int i = rem / N, j = rem % N;
    const float* pi = coords + ((size_t)b * N + i) * 3;
    const float* pj = coords + ((size_t)b * N + j) * 3;
    const float dx = pi[0] - pj[0], dy = pi[1] - pj[1], dz = pi[2] - pj[2];
    float dis = sqrtf(dx * dx + dy * dy + dz * dz + 1e-12f);
    dis = fminf(dis, 30.f);
    int idx = (int)floorf((dis + 1.f) * 0.5f);
    idx = idx < 0 ? 0 : (idx > 15 ? 15 : idx);
    out[(size_t)pid * 128 + c] = (_Float16)(W[idx * 128 + c] + bias[c]);
}

// LN + gated dual projection (pp->ppg, cp->cpg); output channel-major f16 [b][c][rows]
template <typename T>
__global__ void k_gate_proj(const T* __restrict__ X, const float* __restrict__ lng,
                            const float* __restrict__ lnb,
                            const _Float16* __restrict__ WgT, const float* __restrict__ bg,
                            const _Float16* __restrict__ WvT, const float* __restrict__ bv,
                            _Float16* __restrict__ outT, int NRb) {
    __shared__ _Float16 sA[16 * LDL_];
    const int t = threadIdx.x;
    const size_t r0 = (size_t)blockIdx.x * 16;
    ln_tile_to_lds(X, r0, lng, lnb, sA, t);
    __syncthreads();
    const int w = t >> 5, lane = t & 31, hi8 = (lane & 16) ? 8 : 0;
    const int b = (int)(r0 / NRb);
    const size_t rl = r0 % NRb;
    for (int nt = w; nt < 8; nt += 4) {
        const int n0 = nt * 16;
        v8f g = gemm_tile(sA, LDL_, WgT, 128, n0, 128, lane);
        v8f v = gemm_tile(sA, LDL_, WvT, 128, n0, 128, lane);
        const int c = n0 + (lane & 15);
        v8h pk;
#pragma unroll
        for (int rg = 0; rg < 8; ++rg)
            pk[rg] = (_Float16)(sigmoidf_(g[rg] + bg[c]) * (v[rg] + bv[c]));
        *(v8h*)(outT + (size_t)(b * 128 + c) * NRb + rl + hi8) = pk;
    }
}

// LN(z) + {ab1 (wg1/w1), ab2 (wg2/w2), gate (wge)}; einsum-ready transposed f16 outputs
__global__ void k_tri_z(const float* __restrict__ Z, const float* __restrict__ mask,
                        const float* lng, const float* lnb,
                        const _Float16* wg1T, const float* bg1,
                        const _Float16* w1T,  const float* b1,
                        const _Float16* wg2T, const float* bg2,
                        const _Float16* w2T,  const float* b2,
                        const _Float16* wgeT, const float* bge,
                        _Float16* __restrict__ ab1T, _Float16* __restrict__ ab2T,
                        _Float16* __restrict__ gtri) {
    __shared__ _Float16 sA[16 * LDL_];
    const int t = threadIdx.x;
    const size_t r0 = (size_t)blockIdx.x * 16;
    ln_tile_to_lds(Z, r0, lng, lnb, sA, t);
    __syncthreads();
    const int w = t >> 5, lane = t & 31, hi8 = (lane & 16) ? 8 : 0;
    const int bq = (int)(r0 / ((size_t)NP_ * NC_));
    const size_t rem = r0 % ((size_t)NP_ * NC_);
    const int ik = (int)(rem / NC_);   // z row index i (plays "k" in einsum1)
    const int j0 = (int)(rem % NC_);
    for (int task = w; task < 24; task += 4) {
        const int grp = task >> 3;              // 0:ab1 1:ab2 2:gate
        const int n0 = (task & 7) * 16;
        const int c = n0 + (lane & 15);
        if (grp == 0) {
            v8f g = gemm_tile(sA, LDL_, wg1T, 128, n0, 128, lane);
            v8f v = gemm_tile(sA, LDL_, w1T, 128, n0, 128, lane);
#pragma unroll
            for (int rg = 0; rg < 8; ++rg) {
                const int m = rg + hi8;
                float val = sigmoidf_(g[rg] + bg1[c]) * (v[rg] + b1[c]) * mask[r0 + m];
                ab1T[((size_t)(bq * 128 + c) * NC_ + (j0 + m)) * NP_ + ik] = (_Float16)val;
            }
        } else if (grp == 1) {
            v8f g = gemm_tile(sA, LDL_, wg2T, 128, n0, 128, lane);
            v8f v = gemm_tile(sA, LDL_, w2T, 128, n0, 128, lane);
            v8h pk;
#pragma unroll
            for (int rg = 0; rg < 8; ++rg) {
                const int m = rg + hi8;
                pk[rg] = (_Float16)(sigmoidf_(g[rg] + bg2[c]) * (v[rg] + b2[c]) * mask[r0 + m]);
            }
            *(v8h*)(ab2T + ((size_t)(bq * 128 + c) * NP_ + ik) * NC_ + j0 + hi8) = pk;
        } else {
            v8f g = gemm_tile(sA, LDL_, wgeT, 128, n0, 128, lane);
#pragma unroll
            for (int rg = 0; rg < 8; ++rg)
                gtri[(r0 + rg + hi8) * 128 + c] = (_Float16)sigmoidf_(g[rg] + bge[c]);
        }
    }
}

// block[b,i,j,c] = sum_k ppg[b,i,k,c]*ab1[b,k,j,c] + sum_k ab2[b,i,k,c]*cpg[b,j,k,c]
__global__ void k_einsum(const _Float16* __restrict__ ppgT, const _Float16* __restrict__ ab1T,
                         const _Float16* __restrict__ ab2T, const _Float16* __restrict__ cpgT,
                         float* __restrict__ blk) {
    const int t = threadIdx.x, w = t >> 5, lane = t & 31, hi8 = (lane & 16) ? 8 : 0;
    const int tile = blockIdx.x * 4 + w;
    const int jt = tile & 7;
    const int it = (tile >> 3) & 31;
    const int c  = (tile >> 8) & 127;
    const int b  = tile >> 15;
    const int i0 = it * 16, j0 = jt * 16;
    const size_t bc = (size_t)(b * 128 + c);
    const _Float16* A1 = ppgT + (bc * NP_ + i0) * NP_;
    const _Float16* B1 = ab1T + (bc * NC_ + j0) * NP_;
    const _Float16* A2 = ab2T + (bc * NP_ + i0) * NC_;
    const _Float16* B2 = cpgT + (bc * NC_ + j0) * NC_;
    v8f acc = v8f_zero();
    for (int k0 = 0; k0 < NP_; k0 += 32) {
        __builtin_prefetch(A1 + k0 + 128, 0, 1);
        __builtin_prefetch(B1 + k0 + 128, 0, 1);
        acc = wmma32(frag_a(A1, NP_, k0, lane), frag_bT(B1, NP_, 0, k0, lane), acc);
    }
    for (int k0 = 0; k0 < NC_; k0 += 32)
        acc = wmma32(frag_a(A2, NC_, k0, lane), frag_bT(B2, NC_, 0, k0, lane), acc);
    const int j = j0 + (lane & 15);
#pragma unroll
    for (int rg = 0; rg < 8; ++rg) {
        const int i = i0 + rg + hi8;
        blk[((size_t)(b * NP_ + i) * NC_ + j) * 128 + c] = acc[rg];
    }
}

// z += sigmoid(g) * (LN_c(block) @ wo + bo) * mask
__global__ void k_out_proj(const float* __restrict__ blk, const float* lncg, const float* lncb,
                           const _Float16* woT, const float* bo,
                           const _Float16* __restrict__ gtri, const float* __restrict__ mask,
                           float* __restrict__ Z) {
    __shared__ _Float16 sA[16 * LDL_];
    const int t = threadIdx.x;
    const size_t r0 = (size_t)blockIdx.x * 16;
    ln_tile_to_lds(blk, r0, lncg, lncb, sA, t);
    __syncthreads();
    const int w = t >> 5, lane = t & 31, hi8 = (lane & 16) ? 8 : 0;
    for (int nt = w; nt < 8; nt += 4) {
        const int n0 = nt * 16;
        v8f acc = gemm_tile(sA, LDL_, woT, 128, n0, 128, lane);
        const int col = n0 + (lane & 15);
#pragma unroll
        for (int rg = 0; rg < 8; ++rg) {
            const size_t r = r0 + rg + hi8;
            Z[r * 128 + col] += (acc[rg] + bo[col]) * (float)gtri[r * 128 + col] * mask[r];
        }
    }
}

// LN(z) -> q,k (natural f16), v (transposed [row,hc][key]), g (sigmoid)
__global__ void k_attn_proj(const float* __restrict__ Z, const float* lng, const float* lnb,
                            const _Float16* wqT, const _Float16* wkT, const _Float16* wvT,
                            const _Float16* wgT, const float* bg,
                            _Float16* __restrict__ q, _Float16* __restrict__ k,
                            _Float16* __restrict__ vT, _Float16* __restrict__ g) {
    __shared__ _Float16 sA[16 * LDL_];
    const int t = threadIdx.x;
    const size_t r0 = (size_t)blockIdx.x * 16;
    ln_tile_to_lds(Z, r0, lng, lnb, sA, t);
    __syncthreads();
    const int w = t >> 5, lane = t & 31, hi8 = (lane & 16) ? 8 : 0;
    const size_t rowBlock = r0 / NC_;
    const int j0 = (int)(r0 % NC_);
    for (int task = w; task < 32; task += 4) {
        const int mat = task >> 3, n0 = (task & 7) * 16;
        const int col = n0 + (lane & 15);
        const _Float16* WT = (mat == 0) ? wqT : (mat == 1) ? wkT : (mat == 2) ? wvT : wgT;
        v8f acc = gemm_tile(sA, LDL_, WT, 128, n0, 128, lane);
        if (mat == 0 || mat == 1) {
            _Float16* dst = (mat == 0) ? q : k;
#pragma unroll
            for (int rg = 0; rg < 8; ++rg) dst[(r0 + rg + hi8) * 128 + col] = (_Float16)acc[rg];
        } else if (mat == 2) {
            v8h pk;
#pragma unroll
            for (int rg = 0; rg < 8; ++rg) pk[rg] = (_Float16)acc[rg];
            *(v8h*)(vT + ((size_t)rowBlock * 128 + col) * 128 + j0 + hi8) = pk;
        } else {
#pragma unroll
            for (int rg = 0; rg < 8; ++rg)
                g[(r0 + rg + hi8) * 128 + col] = (_Float16)sigmoidf_(acc[rg] + bg[col]);
        }
    }
}

// fused QK^T -> masked softmax -> PV -> gate, per (b, i, head)
__global__ void k_attn_core(const _Float16* __restrict__ q, const _Float16* __restrict__ kk,
                            const _Float16* __restrict__ vT, const _Float16* __restrict__ g,
                            const float* __restrict__ mask, _Float16* __restrict__ o) {
    __shared__ float    sL[4][16][132];
    __shared__ _Float16 sP[4][16][LDL_];
    __shared__ float    sR[4][16];
    const int t = threadIdx.x, w = t >> 5, lane = t & 31, hi8 = (lane & 16) ? 8 : 0;
    const int h = blockIdx.x & 3;
    const size_t rowBlock = blockIdx.x >> 2;     // b*NP + i
    const size_t rbase = rowBlock * NC_;
    const _Float16* qb = q  + rbase * 128 + h * 32;
    const _Float16* kb = kk + rbase * 128 + h * 32;
    const _Float16* vb = vT + (rowBlock * 128 + h * 32) * 128;
    const float* mrow = mask + rbase;
    for (int qt = w; qt < 8; qt += 4) {
        const int q0 = qt * 16;
        for (int kt = 0; kt < 8; ++kt) {       // logits, K = HC = 32 (single WMMA each)
            v16h a = frag_a(qb + (size_t)q0 * 128, 128, 0, lane);
            v16h b = frag_bT(kb, 128, kt * 16, 0, lane);
            v8f acc = wmma32(a, b, v8f_zero());
            const int kcol = kt * 16 + (lane & 15);
            const float bias = 1e9f * (mrow[kcol] - 1.f);
#pragma unroll
            for (int rg = 0; rg < 8; ++rg) sL[w][rg + hi8][kcol] = acc[rg] + bias;
        }
        __syncthreads();
        {   // unnormalized softmax; 2 lanes per row
            const int row = lane >> 1, half = lane & 1, c0 = half * 64;
            float mx = -3.4e38f;
            for (int cc = 0; cc < 64; ++cc) mx = fmaxf(mx, sL[w][row][c0 + cc]);
            mx = fmaxf(mx, __shfl_xor(mx, 1, 32));
            float sm = 0.f;
            for (int cc = 0; cc < 64; ++cc) {
                float e = __expf(sL[w][row][c0 + cc] - mx);
                sm += e;
                sP[w][row][c0 + cc] = (_Float16)e;
            }
            sm += __shfl_xor(sm, 1, 32);
            if (half == 0) sR[w][row] = fast_rcp(sm);
        }
        __syncthreads();
        for (int ct = 0; ct < 2; ++ct) {       // O = P~ @ V, scale by 1/sum, gate
            v8f acc = v8f_zero();
            for (int k0 = 0; k0 < 128; k0 += 32)
                acc = wmma32(frag_a(&sP[w][0][0], LDL_, k0, lane),
                             frag_bT(vb, 128, ct * 16, k0, lane), acc);
            const int col = h * 32 + ct * 16 + (lane & 15);
#pragma unroll
            for (int rg = 0; rg < 8; ++rg) {
                const int m = rg + hi8;
                const size_t r = rbase + q0 + m;
                o[r * 128 + col] = (_Float16)(acc[rg] * sR[w][m] * (float)g[r * 128 + col]);
            }
        }
        __syncthreads();
    }
}

// z += (attn_o @ wf + bf) * mask
__global__ void k_wf_proj(const _Float16* __restrict__ X, const _Float16* wfT, const float* bf,
                          const float* __restrict__ mask, float* __restrict__ Z) {
    __shared__ _Float16 sA[16 * LDL_];
    const int t = threadIdx.x;
    const size_t r0 = (size_t)blockIdx.x * 16;
    for (int idx = t; idx < 16 * 128; idx += 128) {
        int m = idx >> 7, c = idx & 127;
        sA[m * LDL_ + c] = X[(r0 + m) * 128 + c];
    }
    __syncthreads();
    const int w = t >> 5, lane = t & 31, hi8 = (lane & 16) ? 8 : 0;
    for (int nt = w; nt < 8; nt += 4) {
        const int n0 = nt * 16;
        v8f acc = gemm_tile(sA, LDL_, wfT, 128, n0, 128, lane);
        const int col = n0 + (lane & 15);
#pragma unroll
        for (int rg = 0; rg < 8; ++rg) {
            const size_t r = r0 + rg + hi8;
            Z[r * 128 + col] += (acc[rg] + bf[col]) * mask[r];
        }
    }
}

// h = relu(LN(z) @ w1 + b1), f16
__global__ void k_trans1(const float* __restrict__ Z, const float* lng, const float* lnb,
                         const _Float16* w1T, const float* b1, _Float16* __restrict__ H) {
    __shared__ _Float16 sA[16 * LDL_];
    const int t = threadIdx.x;
    const size_t r0 = (size_t)blockIdx.x * 16;
    ln_tile_to_lds(Z, r0, lng, lnb, sA, t);
    __syncthreads();
    const int w = t >> 5, lane = t & 31, hi8 = (lane & 16) ? 8 : 0;
    for (int nt = w; nt < 32; nt += 4) {
        const int n0 = nt * 16;
        v8f acc = gemm_tile(sA, LDL_, w1T, 128, n0, 128, lane);
        const int col = n0 + (lane & 15);
#pragma unroll
        for (int rg = 0; rg < 8; ++rg)
            H[(r0 + rg + hi8) * 512 + col] = (_Float16)fmaxf(acc[rg] + b1[col], 0.f);
    }
}

// z = h @ w2 + b2 (overwrite)
__global__ void k_trans2(const _Float16* __restrict__ H, const _Float16* w2T, const float* b2,
                         float* __restrict__ Z) {
    __shared__ _Float16 sA[16 * 528];
    const int t = threadIdx.x;
    const size_t r0 = (size_t)blockIdx.x * 16;
    for (int idx = t; idx < 16 * 512; idx += 128) {
        int m = idx >> 9, c = idx & 511;
        sA[m * 528 + c] = H[(r0 + m) * 512 + c];
    }
    __syncthreads();
    const int w = t >> 5, lane = t & 31, hi8 = (lane & 16) ? 8 : 0;
    for (int nt = w; nt < 8; nt += 4) {
        const int n0 = nt * 16;
        v8f acc = gemm_tile(sA, 528, w2T, 512, n0, 512, lane);
        const int col = n0 + (lane & 15);
#pragma unroll
        for (int rg = 0; rg < 8; ++rg)
            Z[(r0 + rg + hi8) * 128 + col] = acc[rg] + b2[col];
    }
}

__global__ void k_readout(const float* __restrict__ Z, const float* __restrict__ wt,
                          const float* __restrict__ bt, const float* __restrict__ mask,
                          float* __restrict__ y) {
    const size_t r = (size_t)blockIdx.x * blockDim.x + threadIdx.x;
    if (r >= (size_t)RZ_) return;
    const float* zr = Z + r * 128;
    float s = 0.f;
#pragma unroll 4
    for (int c = 0; c < 128; ++c) s += zr[c] * wt[c];
    y[r] = (s + bt[0]) * mask[r];
}

// =================== host ===================

extern "C" void kernel_launch(void* const* d_in, const int* in_sizes, int n_in,
                              void* d_out, int out_size, void* d_ws, size_t ws_size,
                              hipStream_t stream) {
    const float* protein  = (const float*)d_in[0];
    const float* compound = (const float*)d_in[1];
    const float* z_in     = (const float*)d_in[2];
    const float* z_mask   = (const float*)d_in[3];
    const float* Wp = (const float*)d_in[4];  const float* bp = (const float*)d_in[5];
    const float* Wc = (const float*)d_in[6];  const float* bc = (const float*)d_in[7];
    // tri
    const float* t_lng = (const float*)d_in[8];   const float* t_lnb = (const float*)d_in[9];
    const float* t_lcg = (const float*)d_in[10];  const float* t_lcb = (const float*)d_in[11];
    const float* wg1 = (const float*)d_in[12];    const float* bg1 = (const float*)d_in[13];
    const float* wg2 = (const float*)d_in[14];    const float* bg2 = (const float*)d_in[15];
    const float* w1  = (const float*)d_in[16];    const float* b1  = (const float*)d_in[17];
    const float* w2  = (const float*)d_in[18];    const float* b2  = (const float*)d_in[19];
    const float* wge = (const float*)d_in[20];    const float* bge = (const float*)d_in[21];
    const float* wo  = (const float*)d_in[22];    const float* bo  = (const float*)d_in[23];
    // attn
    const float* a_lng = (const float*)d_in[24];  const float* a_lnb = (const float*)d_in[25];
    const float* wq = (const float*)d_in[26];     const float* wk = (const float*)d_in[27];
    const float* wv = (const float*)d_in[28];     const float* wg = (const float*)d_in[29];
    const float* bg = (const float*)d_in[30];
    const float* wf = (const float*)d_in[31];     const float* bf = (const float*)d_in[32];
    // trans
    const float* r_lng = (const float*)d_in[33];  const float* r_lnb = (const float*)d_in[34];
    const float* tw1 = (const float*)d_in[35];    const float* tb1 = (const float*)d_in[36];
    const float* tw2 = (const float*)d_in[37];    const float* tb2 = (const float*)d_in[38];
    const float* rw  = (const float*)d_in[39];    const float* rb  = (const float*)d_in[40];

    // ---- workspace layout (aliased regions; all transient buffers reused) ----
    char* base = (char*)d_ws;
    size_t off = 0;
    auto take = [&](size_t bytes) { size_t o = off; off = (off + bytes + 255) & ~((size_t)255); return o; };
    const size_t o_pp   = take((size_t)RPP_ * 128 * 2);   // pp f16
    const size_t o_cp   = take((size_t)RCP_ * 128 * 2);   // cp f16
    const size_t o_z    = take((size_t)RZ_  * 128 * 4);   // running z f32
    const size_t o_big  = take((size_t)B_ * 128 * NP_ * NP_ * 2); // ppgT | h(trans) (equal size)
    const size_t o_cpg  = take((size_t)B_ * 128 * NC_ * NC_ * 2);
    const size_t o_b1   = take((size_t)B_ * 128 * NC_ * NP_ * 2); // ab1T | q
    const size_t o_b2   = take((size_t)B_ * 128 * NC_ * NP_ * 2); // ab2T | k
    const size_t o_b3   = take((size_t)B_ * 128 * NC_ * NP_ * 2); // gtri | vT
    const size_t o_blk  = take((size_t)RZ_ * 128 * 4);            // block f32 | (g_attn + attn_o)
    const size_t o_wts  = take((size_t)(5 * 11 * 16384 + 2 * 65536) * 2);
    (void)ws_size; (void)in_sizes; (void)n_in; (void)out_size;

    _Float16* pp   = (_Float16*)(base + o_pp);
    _Float16* cp   = (_Float16*)(base + o_cp);
    float*    zbuf = (float*)(base + o_z);
    _Float16* ppgT = (_Float16*)(base + o_big);
    _Float16* hbuf = (_Float16*)(base + o_big);
    _Float16* cpgT = (_Float16*)(base + o_cpg);
    _Float16* ab1T = (_Float16*)(base + o_b1);   _Float16* qbuf = ab1T;
    _Float16* ab2T = (_Float16*)(base + o_b2);   _Float16* kbuf = ab2T;
    _Float16* gtri = (_Float16*)(base + o_b3);   _Float16* vTb  = gtri;
    float*    blk  = (float*)(base + o_blk);
    _Float16* gatt = (_Float16*)(base + o_blk);
    _Float16* aout = gatt + (size_t)RZ_ * 128;
    _Float16* wbase = (_Float16*)(base + o_wts);
    auto WT = [&](int s, int idx) { return wbase + ((size_t)s * 11 + idx) * 16384; };
    _Float16* tw1T = wbase + (size_t)5 * 11 * 16384;
    _Float16* tw2T = tw1T + 65536;

    // weight transposes (f32 [K,N] -> f16 [N,K])
    auto T = [&](const float* s, _Float16* d, int K, int N) {
        k_transpose_f16<<<(K * N + 255) / 256, 256, 0, stream>>>(s, d, K, N);
    };
    for (int s = 0; s < S_; ++s) {
        const size_t o = (size_t)s * 128 * 128;
        T(wg1 + o, WT(s, 0), 128, 128);  T(w1 + o, WT(s, 1), 128, 128);
        T(wg2 + o, WT(s, 2), 128, 128);  T(w2 + o, WT(s, 3), 128, 128);
        T(wge + o, WT(s, 4), 128, 128);  T(wo + o, WT(s, 5), 128, 128);
        T(wq + o, WT(s, 6), 128, 128);   T(wk + o, WT(s, 7), 128, 128);
        T(wv + o, WT(s, 8), 128, 128);   T(wg + o, WT(s, 9), 128, 128);
        T(wf + o, WT(s, 10), 128, 128);
    }
    T(tw1, tw1T, 128, 512);
    T(tw2, tw2T, 512, 128);

    // pair embeddings + z init
    k_pair_emb<<<B_ * NP_ * NP_, 128, 0, stream>>>(protein, Wp, bp, pp, NP_);
    k_pair_emb<<<B_ * NC_ * NC_, 128, 0, stream>>>(compound, Wc, bc, cp, NC_);
    hipMemcpyAsync(zbuf, z_in, (size_t)RZ_ * 128 * 4, hipMemcpyDeviceToDevice, stream);

    for (int s = 0; s < S_; ++s) {
        const float* lng = t_lng + s * 128;  const float* lnb = t_lnb + s * 128;
        k_gate_proj<_Float16><<<RPP_ / 16, 128, 0, stream>>>(
            pp, lng, lnb, WT(s, 2), bg2 + s * 128, WT(s, 3), b2 + s * 128, ppgT, NP_ * NP_);
        k_gate_proj<_Float16><<<RCP_ / 16, 128, 0, stream>>>(
            cp, lng, lnb, WT(s, 0), bg1 + s * 128, WT(s, 1), b1 + s * 128, cpgT, NC_ * NC_);
        k_tri_z<<<RZ_ / 16, 128, 0, stream>>>(
            zbuf, z_mask, lng, lnb,
            WT(s, 0), bg1 + s * 128, WT(s, 1), b1 + s * 128,
            WT(s, 2), bg2 + s * 128, WT(s, 3), b2 + s * 128,
            WT(s, 4), bge + s * 128, ab1T, ab2T, gtri);
        k_einsum<<<(B_ * 128 * 32 * 8) / 4, 128, 0, stream>>>(ppgT, ab1T, ab2T, cpgT, blk);
        k_out_proj<<<RZ_ / 16, 128, 0, stream>>>(
            blk, t_lcg + s * 128, t_lcb + s * 128, WT(s, 5), bo + s * 128, gtri, z_mask, zbuf);
        k_attn_proj<<<RZ_ / 16, 128, 0, stream>>>(
            zbuf, a_lng + s * 128, a_lnb + s * 128,
            WT(s, 6), WT(s, 7), WT(s, 8), WT(s, 9), bg + s * 128,
            qbuf, kbuf, vTb, gatt);
        k_attn_core<<<B_ * NP_ * HEADS_, 128, 0, stream>>>(qbuf, kbuf, vTb, gatt, z_mask, aout);
        k_wf_proj<<<RZ_ / 16, 128, 0, stream>>>(aout, WT(s, 10), bf + s * 128, z_mask, zbuf);
        k_trans1<<<RZ_ / 16, 128, 0, stream>>>(zbuf, r_lng, r_lnb, tw1T, tb1, hbuf);
        k_trans2<<<RZ_ / 16, 128, 0, stream>>>(hbuf, tw2T, tb2, zbuf);
    }
    k_readout<<<(RZ_ + 255) / 256, 256, 0, stream>>>(zbuf, rw, rb, z_mask, (float*)d_out);
}
